// CausalAttention_2671469658180
// MI455X (gfx1250) — matrix operand order repaired and checked by reading, not compiled, not run on hardware
//
#include <hip/hip_runtime.h>
#include <hip/hip_bf16.h>

typedef __attribute__((ext_vector_type(16))) _Float16 v16h;
typedef __attribute__((ext_vector_type(8)))  _Float16 v8h;
typedef __attribute__((ext_vector_type(8)))  float    v8f;

#define BB   2
#define SS   2048
#define DIMM 2048
#define HH   16
#define DD   128

// ---------------------------------------------------------------------------
// Kernel 0: fp32 -> f16 conversion (8 elements / thread, fully coalesced).
// ---------------------------------------------------------------------------
__global__ __launch_bounds__(256) void cvt_f16(const float* __restrict__ src,
                                               _Float16* __restrict__ dst) {
    const size_t i = ((size_t)blockIdx.x * blockDim.x + threadIdx.x) * 8;
    const float4 a = *(const float4*)(src + i);
    const float4 b = *(const float4*)(src + i + 4);
    v8h o;
    o[0] = (_Float16)a.x; o[1] = (_Float16)a.y;
    o[2] = (_Float16)a.z; o[3] = (_Float16)a.w;
    o[4] = (_Float16)b.x; o[5] = (_Float16)b.y;
    o[6] = (_Float16)b.z; o[7] = (_Float16)b.w;
    *(v8h*)(dst + i) = o;
}

// ---------------------------------------------------------------------------
// Kernel 1: y = x @ W^T on pre-converted f16, fp32 WMMA accumulate.
// Wave tile 64(M) x 64(N): 4 A-frags x 4 B-frags -> 16 WMMAs per K-step
// (1:1 load:WMMA instruction ratio, 33% less L2 operand traffic vs 32x64).
// Block = 8 waves (2 M x 4 N) -> 128(M) x 256(N) tile.  grid.z = {Q,K,V}.
// Epilogue: RoPE (Q,K) via lane-pair shfl, f16 stores in attention layout:
//   Q,K -> [b][h][s][d],  V -> [b][h][d][s].
// ---------------------------------------------------------------------------
__global__ __launch_bounds__(256) void qkv_gemm_rope(
    const _Float16* __restrict__ xh,
    const _Float16* __restrict__ Wqh, const _Float16* __restrict__ Wkh,
    const _Float16* __restrict__ Wvh,
    _Float16* __restrict__ Qh, _Float16* __restrict__ Kh,
    _Float16* __restrict__ Vt)
{
    const int lane  = threadIdx.x & 31;
    const int wave  = threadIdx.x >> 5;          // 0..7
    const int wid_m = wave & 1;                  // 2 M-waves
    const int wid_n = wave >> 1;                 // 4 N-waves
    const int proj  = blockIdx.z;                // 0=Q 1=K 2=V
    const _Float16* W = (proj == 0) ? Wqh : (proj == 1) ? Wkh : Wvh;

    const int n0 = blockIdx.x * 256 + wid_n * 64;
    const int m0 = blockIdx.y * 128 + wid_m * 64;

    // Contiguous k mapping (lane half h holds k = 16h..16h+15, A and B alike) (16-bit A 16x32 / B 32x16)
    const int am  = lane & 15;
    const int akb = (lane >> 4) * 16;
    const int bn  = lane & 15;
    const int bkb = (lane >> 4) * 16;

    const _Float16* arow[4];
    const _Float16* brow[4];
#pragma unroll
    for (int mt = 0; mt < 4; ++mt)
        arow[mt] = xh + (size_t)(m0 + mt * 16 + am) * DIMM + akb;
#pragma unroll
    for (int nt = 0; nt < 4; ++nt)
        brow[nt] = W + (size_t)(n0 + nt * 16 + bn) * DIMM + bkb;

    v8f acc[4][4] = {};

    for (int k0 = 0; k0 < DIMM; k0 += 32) {
        v16h a[4], b[4];
#pragma unroll
        for (int mt = 0; mt < 4; ++mt) {
            v8h lo = *(const v8h*)(arow[mt] + k0);
            v8h hi = *(const v8h*)(arow[mt] + k0 + 8);
            a[mt] = __builtin_shufflevector(lo, hi, 0,1,2,3,4,5,6,7,8,9,10,11,12,13,14,15);
        }
#pragma unroll
        for (int nt = 0; nt < 4; ++nt) b[nt] = *(const v16h*)(brow[nt] + k0);
#pragma unroll
        for (int mt = 0; mt < 4; ++mt)
#pragma unroll
            for (int nt = 0; nt < 4; ++nt)
                acc[mt][nt] = __builtin_amdgcn_wmma_f32_16x16x32_f16(
                    false, a[mt], false, b[nt], (short)0, acc[mt][nt], false, false);
    }

    // Epilogue: C layout -> VGPR r = row r (lanes 0-15) / r+8 (lanes 16-31),
    // column = lane&15 in the 16-wide N tile.  RoPE pair (n, n^1) = lane^1.
    const float rope_c = 0.07195587823f;         // ln(10000)/128
    float theta[4];                              // RoPE theta per N-tile (dj fixed per lane)
#pragma unroll
    for (int nt = 0; nt < 4; ++nt) {
        const int dj = (n0 + nt * 16 + (lane & 15)) & (DD - 1);
        theta[nt] = __expf(-rope_c * (float)(dj & ~1));
    }

#pragma unroll
    for (int mt = 0; mt < 4; ++mt) {
#pragma unroll
        for (int nt = 0; nt < 4; ++nt) {
#pragma unroll
            for (int r = 0; r < 8; ++r) {
                const int row = r + ((lane >> 4) << 3);
                const int gm  = m0 + mt * 16 + row;
                const int bi  = gm >> 11;
                const int si  = gm & (SS - 1);
                const int n   = n0 + nt * 16 + (lane & 15);
                const int h   = n >> 7;
                const int dj  = n & (DD - 1);
                float v = acc[mt][nt][r];
                if (proj < 2) {
                    const float other = __shfl_xor(v, 1, 32);
                    const float freq  = (float)si * theta[nt];
                    float sn, cs;
                    __sincosf(freq, &sn, &cs);
                    v = (dj & 1) ? (other * sn + v * cs)
                                 : (v * cs - other * sn);
                    _Float16* dst = (proj == 0) ? Qh : Kh;
                    dst[(((size_t)bi * HH + h) * SS + si) * DD + dj] = (_Float16)v;
                } else {
                    Vt[(((size_t)bi * HH + h) * DD + dj) * SS + si] = (_Float16)v;
                }
            }
        }
    }
}

// ---------------------------------------------------------------------------
// Kernel 2: causal flash attention. One wave owns a 16-row query tile and
// streams 32-key blocks up to the diagonal. S and PV both on WMMA f16->f32.
// ---------------------------------------------------------------------------
__global__ __launch_bounds__(256) void flash_attn(
    const _Float16* __restrict__ Qh, const _Float16* __restrict__ Kh,
    const _Float16* __restrict__ Vt, float* __restrict__ out)
{
    __shared__ __align__(32) _Float16 ldsP[8][16][40];   // per-wave P staging, padded

    const int lane = threadIdx.x & 31;
    const int wave = threadIdx.x >> 5;
    const int bh   = blockIdx.y;                          // b*16 + h
    const int q0   = (blockIdx.x * 8 + wave) * 16;

    const _Float16* qbase = Qh + (size_t)bh * SS * DD;
    const _Float16* kbase = Kh + (size_t)bh * SS * DD;
    const _Float16* vbase = Vt + (size_t)bh * DD * SS;

    const int am  = lane & 15;
    const int akb = (lane >> 4) * 16;
    const int bn  = lane & 15;
    const int bkb = (lane >> 4) * 16;

    // Q A-fragments for d=128 (4 x K=32)
    v16h qa[4];
    {
        const _Float16* qrow = qbase + (size_t)(q0 + am) * DD;
#pragma unroll
        for (int kk = 0; kk < 4; ++kk) {
            v8h lo = *(const v8h*)(qrow + kk * 32 + akb);
            v8h hi = *(const v8h*)(qrow + kk * 32 + akb + 8);
            qa[kk] = __builtin_shufflevector(lo, hi, 0,1,2,3,4,5,6,7,8,9,10,11,12,13,14,15);
        }
    }

    v8f o[8] = {};
    float mrun[8], lrun[8];
#pragma unroll
    for (int r = 0; r < 8; ++r) { mrun[r] = -1e30f; lrun[r] = 0.0f; }

    const float scale = 0.08838834764831845f;             // 1/sqrt(128)
    const int kend = q0 + 16;

    for (int kt = 0; kt < kend; kt += 32) {
        // ---- S = Q K^T for two 16-key halves ----
        v8f s0 = {}, s1 = {};
        const _Float16* k0row = kbase + (size_t)(kt + bn) * DD;
        const _Float16* k1row = kbase + (size_t)(kt + 16 + bn) * DD;
#pragma unroll
        for (int kk = 0; kk < 4; ++kk) {
            v16h kb0 = *(const v16h*)(k0row + kk * 32 + bkb);
            s0 = __builtin_amdgcn_wmma_f32_16x16x32_f16(
                false, qa[kk], false, kb0, (short)0, s0, false, false);
        }
#pragma unroll
        for (int kk = 0; kk < 4; ++kk) {
            v16h kb1 = *(const v16h*)(k1row + kk * 32 + bkb);
            s1 = __builtin_amdgcn_wmma_f32_16x16x32_f16(
                false, qa[kk], false, kb1, (short)0, s1, false, false);
        }

        float sv0[8], sv1[8];
#pragma unroll
        for (int r = 0; r < 8; ++r) {
            const int rrow = q0 + r + ((lane >> 4) << 3);
            const int c0   = kt + (lane & 15);
            sv0[r] = (c0      <= rrow) ? s0[r] * scale : -1e30f;
            sv1[r] = (c0 + 16 <= rrow) ? s1[r] * scale : -1e30f;
        }

        // ---- online softmax: row reductions across the 16-lane N group ----
#pragma unroll
        for (int r = 0; r < 8; ++r) {
            float mx = fmaxf(sv0[r], sv1[r]);
            mx = fmaxf(mx, __shfl_xor(mx, 1, 32));
            mx = fmaxf(mx, __shfl_xor(mx, 2, 32));
            mx = fmaxf(mx, __shfl_xor(mx, 4, 32));
            mx = fmaxf(mx, __shfl_xor(mx, 8, 32));
            const float mnew  = fmaxf(mrun[r], mx);
            const float alpha = __expf(mrun[r] - mnew);
            const float p0 = __expf(sv0[r] - mnew);
            const float p1 = __expf(sv1[r] - mnew);
            float rs = p0 + p1;
            rs += __shfl_xor(rs, 1, 32);
            rs += __shfl_xor(rs, 2, 32);
            rs += __shfl_xor(rs, 4, 32);
            rs += __shfl_xor(rs, 8, 32);
            lrun[r] = lrun[r] * alpha + rs;
            mrun[r] = mnew;
#pragma unroll
            for (int t = 0; t < 8; ++t) o[t][r] *= alpha;
            const int row = r + ((lane >> 4) << 3);
            const int c   = lane & 15;
            ldsP[wave][row][c]      = (_Float16)p0;       // C->A transpose via LDS
            ldsP[wave][row][16 + c] = (_Float16)p1;
        }

        // ---- P A-fragment (same-wave DS ordering, no barrier needed) ----
        v16h pa;
#pragma unroll
        for (int i = 0; i < 8; ++i) {
            pa[i]     = ldsP[wave][am][akb + i];
            pa[i + 8] = ldsP[wave][am][akb + 8 + i];
        }

        // ---- O += P V  (V^T gives contiguous key runs per lane) ----
#pragma unroll
        for (int t = 0; t < 8; ++t) {
            const _Float16* vrow = vbase + (size_t)(t * 16 + bn) * SS + kt + bkb;
            v16h vb = *(const v16h*)(vrow);
            o[t] = __builtin_amdgcn_wmma_f32_16x16x32_f16(
                false, pa, false, vb, (short)0, o[t], false, false);
        }
    }

    // ---- normalize and store out[b][s][h*128+dj] (fp32) ----
    const int bi = bh >> 4;
    const int h  = bh & 15;
#pragma unroll
    for (int t = 0; t < 8; ++t) {
#pragma unroll
        for (int r = 0; r < 8; ++r) {
            const int row = r + ((lane >> 4) << 3);
            const float val = o[t][r] / lrun[r];
            out[((size_t)bi * SS + (q0 + row)) * DIMM + h * DD + t * 16 + (lane & 15)] = val;
        }
    }
}

extern "C" void kernel_launch(void* const* d_in, const int* in_sizes, int n_in,
                              void* d_out, int out_size, void* d_ws, size_t ws_size,
                              hipStream_t stream) {
    const float* x  = (const float*)d_in[0];
    const float* Wq = (const float*)d_in[1];
    const float* Wk = (const float*)d_in[2];
    const float* Wv = (const float*)d_in[3];
    float* out = (float*)d_out;

    const size_t per  = (size_t)BB * HH * SS * DD;   // 8M f16 (Q/K/V tensors)
    const size_t xcnt = (size_t)BB * SS * DIMM;      // 8M f16 (x)
    const size_t wcnt = (size_t)DIMM * DIMM;         // 4M f16 (each W)

    _Float16* Qh  = (_Float16*)d_ws;
    _Float16* Kh  = Qh + per;
    _Float16* Vt  = Kh + per;
    _Float16* xh  = Vt + per;
    _Float16* Wqh = xh + xcnt;
    _Float16* Wkh = Wqh + wcnt;
    _Float16* Wvh = Wkh + wcnt;

    // fp32 -> f16 pre-conversion (memory-bound, one pass)
    cvt_f16<<<dim3((unsigned)(xcnt / 2048)), dim3(256), 0, stream>>>(x, xh);
    cvt_f16<<<dim3((unsigned)(wcnt / 2048)), dim3(256), 0, stream>>>(Wq, Wqh);
    cvt_f16<<<dim3((unsigned)(wcnt / 2048)), dim3(256), 0, stream>>>(Wk, Wkh);
    cvt_f16<<<dim3((unsigned)(wcnt / 2048)), dim3(256), 0, stream>>>(Wv, Wvh);

    // QKV projections + RoPE + layout transform: grid.z = {Q,K,V}
    qkv_gemm_rope<<<dim3(DIMM / 256, (BB * SS) / 128, 3), dim3(256), 0, stream>>>(
        xh, Wqh, Wkh, Wvh, Qh, Kh, Vt);

    // Causal flash attention: 128 query tiles per head, 8 waves per block
    flash_attn<<<dim3(SS / 16 / 8, BB * HH), dim3(256), 0, stream>>>(
        Qh, Kh, Vt, out);
}
